// VQ_MLP_MCQ_46643344834662
// MI455X (gfx1250) — matrix operand order repaired and checked by reading, not compiled, not run on hardware
//
#include <hip/hip_runtime.h>
#include <hip/hip_bf16.h>
#include <stdint.h>

// ---------------------------------------------------------------------------
// VQ-MLP pipeline for MI455X (gfx1250, wave32, WMMA).
//   x(8192x1152) -> GEMM1+GELU -> h1(8192x4608) -> GEMM2 -> z(8192x256)
//   -> VQ (4 codebooks x 2048 codes x 64 dims) -> zq
//   -> GEMM3+GELU -> h3(8192x16384) -> GEMM4 -> x_vq(8192x4096)
// GEMMs: bf16 v_wmma_f32_16x16x32_bf16, f32 accum, double-buffered LDS tiles
// filled by GLOBAL_LOAD_ASYNC_TO_LDS_B128 (ASYNCcnt) when available.
// ---------------------------------------------------------------------------

typedef __attribute__((ext_vector_type(16))) __bf16 v16bf;
typedef __attribute__((ext_vector_type(8)))  __bf16 bf16x8;
typedef __attribute__((ext_vector_type(8)))  float  v8f;
typedef __attribute__((ext_vector_type(4)))  int    v4i;

#define AS(n) __attribute__((address_space(n)))

#if defined(__gfx1250__) && __has_builtin(__builtin_amdgcn_global_load_async_to_lds_b128)
#define HAVE_ASYNC_LDS 1
#else
#define HAVE_ASYNC_LDS 0
#endif

__device__ __forceinline__ void wait_asynccnt0() {
#if __has_builtin(__builtin_amdgcn_s_wait_asynccnt)
  __builtin_amdgcn_s_wait_asynccnt(0);
#else
  asm volatile("s_wait_asynccnt 0" ::: "memory");
#endif
}

#define TOKENS   8192
#define D_IN     1152
#define D_H1     4608
#define D_E      256
#define NCB      4
#define KCODES   2048
#define DPC      64
#define D_H3     16384
#define D_OUT    4096

// ---------------------------------------------------------------------------
// f32 -> bf16 elementwise convert (for activations x)
// ---------------------------------------------------------------------------
__global__ void cvt_f32_bf16_kernel(const float* __restrict__ in,
                                    __bf16* __restrict__ out, long n) {
  long i = (long)blockIdx.x * blockDim.x + threadIdx.x;
  if (i < n) out[i] = (__bf16)in[i];
}

// ---------------------------------------------------------------------------
// f32 (KxN) -> bf16 transposed (NxK), tiled through LDS. K,N multiples of 32.
// ---------------------------------------------------------------------------
__global__ void transpose_cvt_kernel(const float* __restrict__ W,
                                     __bf16* __restrict__ Wt, int K, int N) {
  __shared__ float t[32][33];
  const int nx = blockIdx.x * 32;  // N tile
  const int ky = blockIdx.y * 32;  // K tile
  const int lx = threadIdx.x;      // 0..31
  const int ly = threadIdx.y;      // 0..7
#pragma unroll
  for (int i = 0; i < 4; ++i) {
    int r = ly + i * 8;
    t[r][lx] = W[(size_t)(ky + r) * N + nx + lx];
  }
  __syncthreads();
#pragma unroll
  for (int i = 0; i < 4; ++i) {
    int r = ly + i * 8;
    Wt[(size_t)(nx + r) * K + ky + lx] = (__bf16)t[lx][r];
  }
}

// ---------------------------------------------------------------------------
// Zero-init counts + commitment accumulator (graph-capture safe memset).
// ---------------------------------------------------------------------------
__global__ void zero_kernel(int* __restrict__ counts, float* __restrict__ commit,
                            int n) {
  int i = blockIdx.x * blockDim.x + threadIdx.x;
  if (i < n) counts[i] = 0;
  if (i == 0) commit[0] = 0.0f;
}

// ---------------------------------------------------------------------------
// Tiled bf16 WMMA GEMM: D = act(A(MxK) * B^T(NxK)^T + bias)
// Block tile 128x128, K-step 32, 256 threads = 8 waves in a 2x4 grid;
// each wave computes 4x2 16x16 WMMA subtiles (64x32 patch).
// Double-buffered LDS, async global->LDS DMA overlapped with WMMA.
// ---------------------------------------------------------------------------
#define BM 128
#define BN 128
#define BKT 32
#define LDT 40  // padded LDS row stride in bf16 elements (80B, 16B-aligned)

__device__ __forceinline__ float gelu_exact(float v) {
  return 0.5f * v * (1.0f + erff(v * 0.70710678118654752f));
}

template <bool GELU, bool F32OUT>
__global__ __launch_bounds__(256)
void gemm_bf16_kernel(const __bf16* __restrict__ A,   // M x K row-major
                      const __bf16* __restrict__ Bt,  // N x K row-major (B^T)
                      const float* __restrict__ bias, // N
                      __bf16* __restrict__ outb,      // M x N (if !F32OUT)
                      float* __restrict__ outf,       // M x N (if F32OUT)
                      int M, int N, int K) {
  __shared__ __align__(16) __bf16 As[2][BM * LDT];
  __shared__ __align__(16) __bf16 Bs[2][BN * LDT];

  const int tid  = threadIdx.x;
  const int lane = tid & 31;
  const int wid  = tid >> 5;
  const int wm   = wid >> 2;       // 0..1
  const int wn   = wid & 3;        // 0..3
  const int m0   = blockIdx.y * BM;
  const int n0   = blockIdx.x * BN;
  const int lrow = tid >> 2;       // 0..63 staging row
  const int lcol = (tid & 3) * 8;  // staging col chunk (8 bf16 = 16B)

  const int nlane = lane & 15;
  const int halfk = (lane < 16) ? 0 : 8;   // A-fragment K offset
  const int kb    = (lane < 16) ? 0 : 8;   // B-fragment K base
  const int am    = wm * 64 + nlane;       // A-fragment row base (+= mi*16)
  const int bn    = wn * 32 + nlane;       // B-fragment col base (+= ni*16)

  v8f acc[4][2];
  const v8f vzero = {0.f, 0.f, 0.f, 0.f, 0.f, 0.f, 0.f, 0.f};
#pragma unroll
  for (int i = 0; i < 4; ++i)
#pragma unroll
    for (int j = 0; j < 2; ++j) acc[i][j] = vzero;

#if HAVE_ASYNC_LDS
  // -------- async DMA staging (no VGPR round-trip, ASYNCcnt-tracked) -------
  auto stage = [&](int b, int k0) {
#pragma unroll
    for (int it = 0; it < 2; ++it) {
      const int r = lrow + it * 64;
      __builtin_amdgcn_global_load_async_to_lds_b128(
          (AS(1) v4i*)(A + (size_t)(m0 + r) * K + k0 + lcol),
          (AS(3) v4i*)&As[b][r * LDT + lcol], 0, 0);
      __builtin_amdgcn_global_load_async_to_lds_b128(
          (AS(1) v4i*)(Bt + (size_t)(n0 + r) * K + k0 + lcol),
          (AS(3) v4i*)&Bs[b][r * LDT + lcol], 0, 0);
    }
  };
  stage(0, 0);
  int buf = 0;
#endif

  for (int k0 = 0; k0 < K; k0 += BKT) {
#if HAVE_ASYNC_LDS
    wait_asynccnt0();
    __syncthreads();
    if (k0 + BKT < K) stage(buf ^ 1, k0 + BKT);  // prefetch next tile (DMA)
    const __bf16* Asb = As[buf];
    const __bf16* Bsb = Bs[buf];
#else
    // ---- fallback: load-all then store-all to keep 4 loads in flight ----
    uint4 av[2], bv[2];
#pragma unroll
    for (int it = 0; it < 2; ++it) {
      const int r = lrow + it * 64;
      av[it] = *(const uint4*)(A + (size_t)(m0 + r) * K + k0 + lcol);
      bv[it] = *(const uint4*)(Bt + (size_t)(n0 + r) * K + k0 + lcol);
    }
#pragma unroll
    for (int it = 0; it < 2; ++it) {
      const int r = lrow + it * 64;
      *(uint4*)&As[0][r * LDT + lcol] = av[it];
      *(uint4*)&Bs[0][r * LDT + lcol] = bv[it];
    }
    __syncthreads();
    const __bf16* Asb = As[0];
    const __bf16* Bsb = Bs[0];
#endif

    // ---- build fragments from LDS per documented wave32 WMMA layouts ----
    v16bf afrag[4], bfrag[2];
#pragma unroll
    for (int mi = 0; mi < 4; ++mi) {
      const int m = am + mi * 16;
      union { v16bf v; bf16x8 h[2]; } u;
      u.h[0] = *(const bf16x8*)&Asb[m * LDT + halfk];       // K = halfk..+7
      u.h[1] = *(const bf16x8*)&Asb[m * LDT + 16 + halfk];  // K = 16+halfk..+7
      afrag[mi] = u.v;
    }
#pragma unroll
    for (int ni = 0; ni < 2; ++ni) {
      const int n = bn + ni * 16;
      union { v16bf v; bf16x8 h[2]; } u;
      u.h[0] = *(const bf16x8*)&Bsb[n * LDT + kb];          // K = kb..kb+7
      u.h[1] = *(const bf16x8*)&Bsb[n * LDT + kb + 16];     // K = kb+16..+23
      bfrag[ni] = u.v;
    }

    // ---- 8 WMMAs per wave per K-step ----
#pragma unroll
    for (int mi = 0; mi < 4; ++mi)
#pragma unroll
      for (int ni = 0; ni < 2; ++ni)
        acc[mi][ni] = __builtin_amdgcn_wmma_f32_16x16x32_bf16(
            false, afrag[mi], false, bfrag[ni], (short)0, acc[mi][ni],
            false, false);

#if HAVE_ASYNC_LDS
    buf ^= 1;
#else
    __syncthreads();
#endif
  }

  // ---- epilogue: bias (+GELU), per documented C/D layout (lane=col, vgpr=row)
  const int moff = (lane < 16) ? 0 : 8;
#pragma unroll
  for (int mi = 0; mi < 4; ++mi) {
#pragma unroll
    for (int ni = 0; ni < 2; ++ni) {
      const int n  = n0 + wn * 32 + ni * 16 + nlane;
      const float bv = bias[n];
#pragma unroll
      for (int r = 0; r < 8; ++r) {
        const int m = m0 + wm * 64 + mi * 16 + moff + r;
        float v = acc[mi][ni][r] + bv;
        if (GELU) v = gelu_exact(v);
        const size_t o = (size_t)m * N + n;
        if (F32OUT) outf[o] = v;
        else        outb[o] = (__bf16)v;
      }
    }
  }
}

// ---------------------------------------------------------------------------
// VQ: one block per token. argmin over 2048 codes per codebook, write idx
// (as float into d_out), zq (bf16), histogram counts, commitment partial sum.
// ---------------------------------------------------------------------------
__global__ __launch_bounds__(256)
void vq_kernel(const float* __restrict__ z,       // TOKENS x 256
               const float* __restrict__ cb,      // 4 x 2048 x 64
               __bf16* __restrict__ zq,           // TOKENS x 256
               float* __restrict__ idx_out,       // TOKENS x 4 (as float)
               int* __restrict__ counts,          // 4 x 2048
               float* __restrict__ commit_accum) {
  const int t   = blockIdx.x;
  const int tid = threadIdx.x;
  __shared__ float zs[256];
  __shared__ float rd[256];
  __shared__ int   ri[256];
  __shared__ float block_commit;
  __shared__ int   kstar_s;

  zs[tid] = z[(size_t)t * 256 + tid];
  if (tid == 0) block_commit = 0.0f;
  __syncthreads();

  for (int c = 0; c < NCB; ++c) {
    const float* zc = &zs[c * DPC];
    float bd = 3.4e38f;
    int   bi = 0x7FFFFFFF;
#pragma unroll
    for (int j = 0; j < KCODES / 256; ++j) {
      const int k = tid + j * 256;
      const float* e = &cb[((size_t)c * KCODES + k) * DPC];
      float d = 0.0f;
#pragma unroll
      for (int dd = 0; dd < DPC; ++dd) {
        const float df = zc[dd] - e[dd];
        d = fmaf(df, df, d);
      }
      if (d < bd || (d == bd && k < bi)) { bd = d; bi = k; }
    }
    rd[tid] = bd; ri[tid] = bi;
    __syncthreads();
    for (int s = 128; s > 0; s >>= 1) {
      if (tid < s) {
        const float od = rd[tid + s];
        const int   oi = ri[tid + s];
        if (od < rd[tid] || (od == rd[tid] && oi < ri[tid])) {
          rd[tid] = od; ri[tid] = oi;
        }
      }
      __syncthreads();
    }
    if (tid == 0) {
      kstar_s = ri[0];
      idx_out[(size_t)t * NCB + c] = (float)ri[0];
      atomicAdd(&counts[c * KCODES + ri[0]], 1);
    }
    __syncthreads();
    const int kstar = kstar_s;
    if (tid < DPC) {
      const float e = cb[((size_t)c * KCODES + kstar) * DPC + tid];
      zq[(size_t)t * 256 + c * DPC + tid] = (__bf16)e;
      const float df = zc[tid] - e;
      atomicAdd(&block_commit, df * df);  // ds_add_f32
    }
    __syncthreads();
  }
  if (tid == 0) atomicAdd(commit_accum, block_commit);
}

// ---------------------------------------------------------------------------
// Entropy + scalar finalize (total, commitment, entropy).
// ---------------------------------------------------------------------------
__global__ __launch_bounds__(256)
void finalize_kernel(const int* __restrict__ counts,
                     const float* __restrict__ commit_accum,
                     float* __restrict__ out_scalars) {
  const int tid = threadIdx.x;
  __shared__ float red[256];
  float ent = 0.0f;
  for (int c = 0; c < NCB; ++c)
    for (int k = tid; k < KCODES; k += 256) {
      const float p = (float)counts[c * KCODES + k] / (float)TOKENS;
      ent += -p * logf(p + 1e-10f);
    }
  red[tid] = ent;
  __syncthreads();
  for (int s = 128; s > 0; s >>= 1) {
    if (tid < s) red[tid] += red[tid + s];
    __syncthreads();
  }
  if (tid == 0) {
    const float entropy = red[0] / (float)NCB;
    const float commit  = 0.25f * commit_accum[0] / (float)(TOKENS * 256);
    out_scalars[0] = commit;   // total
    out_scalars[1] = commit;   // commitment
    out_scalars[2] = entropy;  // entropy
  }
}

// ---------------------------------------------------------------------------
// Host-side launch
// ---------------------------------------------------------------------------
extern "C" void kernel_launch(void* const* d_in, const int* in_sizes, int n_in,
                              void* d_out, int out_size, void* d_ws,
                              size_t ws_size, hipStream_t stream) {
  const float* x  = (const float*)d_in[0];
  const float* W1 = (const float*)d_in[1];
  const float* b1 = (const float*)d_in[2];
  const float* W2 = (const float*)d_in[3];
  const float* b2 = (const float*)d_in[4];
  const float* cb = (const float*)d_in[5];
  const float* W3 = (const float*)d_in[6];
  const float* b3 = (const float*)d_in[7];
  const float* W4 = (const float*)d_in[8];
  const float* b4 = (const float*)d_in[9];

  uint8_t* ws = (uint8_t*)d_ws;
  // workspace layout (all offsets 256B-aligned)
  __bf16* xb   = (__bf16*)(ws + 0);                    // 8192x1152
  __bf16* w1t  = (__bf16*)(ws + 18874368ull);          // 4608x1152
  __bf16* w2t  = (__bf16*)(ws + 29491200ull);          // 256x4608
  __bf16* w3t  = (__bf16*)(ws + 31850496ull);          // 16384x256
  __bf16* w4t  = (__bf16*)(ws + 40239104ull);          // 4096x16384
  __bf16* h1b  = (__bf16*)(ws + 174456832ull);         // 8192x4608
  float*  zf   = (float*) (ws + 249954304ull);         // 8192x256 f32
  __bf16* zqb  = (__bf16*)(ws + 258342912ull);         // 8192x256
  __bf16* h3b  = (__bf16*)(ws + 262537216ull);         // 8192x16384
  int*    cnts = (int*)   (ws + 530972672ull);         // 4x2048
  float*  cacc = (float*) (ws + 531005440ull);         // 1

  float* out       = (float*)d_out;
  float* out_xvq   = out;                                      // 8192x4096
  float* out_idx   = out + (size_t)TOKENS * D_OUT;             // 8192x4
  float* out_scal  = out_idx + (size_t)TOKENS * NCB;           // 3 scalars

  const dim3 tb(32, 8);

  // 1) convert x -> bf16; transpose+convert weights -> bf16 NxK
  {
    long n = (long)TOKENS * D_IN;
    cvt_f32_bf16_kernel<<<(unsigned)((n + 255) / 256), 256, 0, stream>>>(x, xb, n);
  }
  transpose_cvt_kernel<<<dim3(D_H1 / 32, D_IN / 32), tb, 0, stream>>>(W1, w1t, D_IN, D_H1);
  transpose_cvt_kernel<<<dim3(D_E / 32, D_H1 / 32), tb, 0, stream>>>(W2, w2t, D_H1, D_E);
  transpose_cvt_kernel<<<dim3(D_H3 / 32, D_E / 32), tb, 0, stream>>>(W3, w3t, D_E, D_H3);
  transpose_cvt_kernel<<<dim3(D_OUT / 32, D_H3 / 32), tb, 0, stream>>>(W4, w4t, D_H3, D_OUT);

  // 2) zero VQ state (counts + commitment accumulator)
  zero_kernel<<<(NCB * KCODES + 255) / 256, 256, 0, stream>>>(cnts, cacc, NCB * KCODES);

  // 3) GEMM1: h1 = gelu(x@W1 + b1)   [8192 x 4608]
  gemm_bf16_kernel<true, false><<<dim3(D_H1 / BN, TOKENS / BM), 256, 0, stream>>>(
      xb, w1t, b1, h1b, nullptr, TOKENS, D_H1, D_IN);

  // 4) GEMM2: z = h1@W2 + b2         [8192 x 256], f32 out for VQ
  gemm_bf16_kernel<false, true><<<dim3(D_E / BN, TOKENS / BM), 256, 0, stream>>>(
      h1b, w2t, b2, nullptr, zf, TOKENS, D_E, D_H1);

  // 5) VQ: argmin, idx, zq, counts, commitment
  vq_kernel<<<TOKENS, 256, 0, stream>>>(zf, cb, zqb, out_idx, cnts, cacc);

  // 6) GEMM3: h3 = gelu(zq@W3 + b3)  [8192 x 16384]
  gemm_bf16_kernel<true, false><<<dim3(D_H3 / BN, TOKENS / BM), 256, 0, stream>>>(
      zqb, w3t, b3, h3b, nullptr, TOKENS, D_H3, D_E);

  // 7) GEMM4: x_vq = h3@W4 + b4      [8192 x 4096], f32 out -> d_out
  gemm_bf16_kernel<false, true><<<dim3(D_OUT / BN, TOKENS / BM), 256, 0, stream>>>(
      h3b, w4t, b4, nullptr, out_xvq, TOKENS, D_OUT, D_H3);

  // 8) entropy + scalars
  finalize_kernel<<<1, 256, 0, stream>>>(cnts, cacc, out_scal);
}
